// Net_15857019256870
// MI455X (gfx1250) — compile-verified, matrix operand order checked by reading, not run on hardware
//
#include <hip/hip_runtime.h>
#include <hip/hip_bf16.h>

// ---------------------------------------------------------------------------
// DGCNN classifier for MI455X (gfx1250, wave32, WMMA f16->f32 16x16x32 tiles)
// B=8 clouds, N=4096 points, K=20 neighbors, OUT=40 classes
// ---------------------------------------------------------------------------

#define BB   8
#define NN   4096
#define KK   20
#define OUTC 40

typedef __attribute__((ext_vector_type(16))) _Float16 v16h;
typedef __attribute__((ext_vector_type(8)))  _Float16 v8h;
typedef __attribute__((ext_vector_type(8)))  float    v8f;

// D = A(16x32 f16) x B(32x16 f16) + C(16x16 f32)
__device__ __forceinline__ v8f wmma32(v16h a, v16h b, v8f c) {
  return __builtin_amdgcn_wmma_f32_16x16x32_f16(false, a, false, b, (short)0, c,
                                                false, false);
}

// A-fragment loader, row-major f16 src with leading dim ld (ld % 8 == 0,
// 16B-aligned rows). ISA 7.12.2: lane L holds M=L%16; K base=(L>>4)*8+kbase;
// elements {base..base+7, base+16..base+23} -> two aligned v8h loads.
__device__ __forceinline__ v16h load_a(const _Float16* src, int ld, int row0, int kbase) {
  int lane = threadIdx.x & 31;
  const _Float16* p = src + (size_t)(row0 + (lane & 15)) * ld + kbase + ((lane >> 4) << 3);
  v8h lo = *(const v8h*)p;
  v8h hi = *(const v8h*)(p + 16);
  return __builtin_shufflevector(lo, hi, 0, 1, 2, 3, 4, 5, 6, 7,
                                 8, 9, 10, 11, 12, 13, 14, 15);
}

// B-fragment loader from fragment-packed weights: fragment f, lane L's 16
// halves contiguous at ((f*32+L)*16). One aligned 32-byte load.
__device__ __forceinline__ v16h load_bp(const _Float16* wp, int frag) {
  int lane = threadIdx.x & 31;
  return *(const v16h*)(wp + (((size_t)frag * 32 + lane) << 4));
}

// ---------------------------------------------------------------------------
// Kernel 0: fold BN into weights, convert to f16, and write each weight in
// WMMA-fragment-packed order. For W[K][N] (nks=K/32): element e of lane L of
// fragment f=(nt*nks+ks) is W[k][n], n = nt*16+(L&15),
// k = ks*32 + ((L>>4)<<3) + (e&7) + (e>=8 ? 16 : 0).
// ---------------------------------------------------------------------------
__device__ __forceinline__ void unpack_frag(int id, int nks, int& k, int& n) {
  int e    = id & 15;
  int lane = (id >> 4) & 31;
  int f    = id >> 9;
  int ks   = f % nks;
  int nt   = f / nks;
  n = nt * 16 + (lane & 15);
  k = ks * 32 + ((lane >> 4) << 3) + (e & 7) + ((e >= 8) ? 16 : 0);
}

__global__ void prep_kernel(
    const float* c1w1, const float* c1b1, const float* c1g1, const float* c1be1,
    const float* c1w2, const float* c1b2, const float* c1g2, const float* c1be2,
    const float* c1w3, const float* c1b3,
    const float* c2w,  const float* c2b,
    const float* lw,   const float* lb,
    _Float16* w1p, float* b1f, _Float16* w2h, float* b2f,
    _Float16* w3h, float* b3f, _Float16* c2wh, float* c2bf,
    _Float16* lwh, float* lbf) {
  int id = blockIdx.x * blockDim.x + threadIdx.x;
  const float inv = rsqrtf(1.f + 1e-5f);
  int k, n;
  if (id < 32 * 64) {                      // w1: K=6 padded to 32, N=64, nks=1
    unpack_frag(id, 1, k, n);
    float s = c1g1[n] * inv;
    w1p[id] = (_Float16)(k < 6 ? c1w1[k * 64 + n] * s : 0.f);
    return;
  }
  id -= 32 * 64;
  if (id < 64) { float s = c1g1[id] * inv; b1f[id] = c1b1[id] * s + c1be1[id]; return; }
  id -= 64;
  if (id < 64 * 64) {                      // w2: 64x64, nks=2
    unpack_frag(id, 2, k, n);
    w2h[id] = (_Float16)(c1w2[k * 64 + n] * c1g2[n] * inv);
    return;
  }
  id -= 64 * 64;
  if (id < 64) { b2f[id] = c1b2[id] * c1g2[id] * inv + c1be2[id]; return; }
  id -= 64;
  if (id < 64 * 64) {                      // w3: 64x64, nks=2
    unpack_frag(id, 2, k, n);
    w3h[id] = (_Float16)c1w3[k * 64 + n];
    return;
  }
  id -= 64 * 64;
  if (id < 64) { b3f[id] = c1b3[id]; return; }
  id -= 64;
  if (id < 128 * 128) {                    // c2w: 128x128, nks=4
    unpack_frag(id, 4, k, n);
    c2wh[id] = (_Float16)c2w[k * 128 + n];
    return;
  }
  id -= 128 * 128;
  if (id < 128) { c2bf[id] = c2b[id]; return; }
  id -= 128;
  if (id < 192 * 1024) {                   // lin1: 192x1024, nks=6
    unpack_frag(id, 6, k, n);
    lwh[id] = (_Float16)lw[k * 1024 + n];
    return;
  }
  id -= 192 * 1024;
  if (id < 1024) { lbf[id] = lb[id]; return; }
}

// ---------------------------------------------------------------------------
// Kernel 1: brute-force kNN in 3D (scalar; D=3 is not WMMA material)
// ---------------------------------------------------------------------------
__global__ __launch_bounds__(256) void knn3_kernel(const float* pos, int* idx1) {
  __shared__ float tile[256 * 3];
  int blk = blockIdx.x;                // 128 blocks, 16 per batch
  int b   = blk >> 4;
  int tid = threadIdx.x;
  int n   = ((blk & 15) << 8) + tid;   // local point id
  size_t g = ((size_t)b << 12) + n;
  float x0 = pos[g * 3], x1 = pos[g * 3 + 1], x2 = pos[g * 3 + 2];
  float bd[KK]; int bi[KK];
  for (int k = 0; k < KK; ++k) { bd[k] = 3.402823e38f; bi[k] = 0; }
  for (int t0 = 0; t0 < NN; t0 += 256) {
    size_t cg = ((size_t)b << 12) + t0 + tid;
    tile[tid * 3 + 0] = pos[cg * 3 + 0];
    tile[tid * 3 + 1] = pos[cg * 3 + 1];
    tile[tid * 3 + 2] = pos[cg * 3 + 2];
    __syncthreads();
    for (int c = 0; c < 256; ++c) {
      float d0 = x0 - tile[c * 3], d1 = x1 - tile[c * 3 + 1], d2 = x2 - tile[c * 3 + 2];
      float d  = d0 * d0 + d1 * d1 + d2 * d2;
      if (d < bd[KK - 1]) {
        int p = KK - 1;
        while (p > 0 && d < bd[p - 1]) { bd[p] = bd[p - 1]; bi[p] = bi[p - 1]; --p; }
        bd[p] = d; bi[p] = t0 + c;
      }
    }
    __syncthreads();
  }
  for (int k = 0; k < KK; ++k) idx1[g * KK + k] = bi[k];
}

// ---------------------------------------------------------------------------
// Kernel 2: EdgeConv1 — one wave per point; 20 edges padded to 32 rows.
// MLP 6(pad32)->64->64->64 via WMMA, BN pre-folded, max over K=20.
// ---------------------------------------------------------------------------
__global__ __launch_bounds__(32) void edgeconv1_kernel(
    const float* pos, const int* idx1,
    const _Float16* w1, const float* b1f,
    const _Float16* w2, const float* b2f,
    const _Float16* w3, const float* b3f,
    _Float16* x1h, _Float16* xcat, float* sq1) {
  __shared__ __align__(32) _Float16 ebuf[32 * 32];
  __shared__ __align__(32) _Float16 abuf[32 * 64];
  __shared__ __align__(32) _Float16 bbuf[32 * 64];
  __shared__ float    fbuf[32 * 64];
  __shared__ float    cm[64];

  int p = blockIdx.x;            // global point 0..32767
  int b = p >> 12;
  int lane = threadIdx.x;

  for (int i = lane; i < 32 * 32; i += 32) ebuf[i] = (_Float16)0.f;
  __syncthreads();
  if (lane < KK) {
    int j   = idx1[(size_t)p * KK + lane];
    size_t gj = ((size_t)b << 12) + j;
    float a0 = pos[(size_t)p * 3], a1 = pos[(size_t)p * 3 + 1], a2 = pos[(size_t)p * 3 + 2];
    float j0 = pos[gj * 3], j1 = pos[gj * 3 + 1], j2 = pos[gj * 3 + 2];
    _Float16* e = &ebuf[lane * 32];
    e[0] = (_Float16)a0;        e[1] = (_Float16)a1;        e[2] = (_Float16)a2;
    e[3] = (_Float16)(j0 - a0); e[4] = (_Float16)(j1 - a1); e[5] = (_Float16)(j2 - a2);
  }
  __syncthreads();

  int col16 = lane & 15, rb8 = (lane >> 4) << 3;

  // Layer 1: [32x32] x [32x64]   (nks=1, frag = nt)
  for (int t = 0; t < 2; ++t) {
    v16h a = load_a(ebuf, 32, t * 16, 0);
    for (int nt = 0; nt < 4; ++nt) {
      v8f c = {};
      c = wmma32(a, load_bp(w1, nt), c);
      int colc = nt * 16 + col16;
      float bias = b1f[colc];
#pragma unroll
      for (int r = 0; r < 8; ++r) {
        float v = c[r] + bias;
        abuf[(t * 16 + rb8 + r) * 64 + colc] = (_Float16)(v > 0.f ? v : 0.f);
      }
    }
  }
  __syncthreads();

  // Layer 2: [32x64] x [64x64]   (nks=2, frag = nt*2+ks)
  for (int t = 0; t < 2; ++t) {
    v16h a0 = load_a(abuf, 64, t * 16, 0);
    v16h a1 = load_a(abuf, 64, t * 16, 32);
    for (int nt = 0; nt < 4; ++nt) {
      v8f c = {};
      c = wmma32(a0, load_bp(w2, nt * 2 + 0), c);
      c = wmma32(a1, load_bp(w2, nt * 2 + 1), c);
      int colc = nt * 16 + col16;
      float bias = b2f[colc];
#pragma unroll
      for (int r = 0; r < 8; ++r) {
        float v = c[r] + bias;
        bbuf[(t * 16 + rb8 + r) * 64 + colc] = (_Float16)(v > 0.f ? v : 0.f);
      }
    }
  }
  __syncthreads();

  // Layer 3 (plain linear): [32x64] x [64x64] -> f32
  for (int t = 0; t < 2; ++t) {
    v16h a0 = load_a(bbuf, 64, t * 16, 0);
    v16h a1 = load_a(bbuf, 64, t * 16, 32);
    for (int nt = 0; nt < 4; ++nt) {
      v8f c = {};
      c = wmma32(a0, load_bp(w3, nt * 2 + 0), c);
      c = wmma32(a1, load_bp(w3, nt * 2 + 1), c);
      int colc = nt * 16 + col16;
      float bias = b3f[colc];
#pragma unroll
      for (int r = 0; r < 8; ++r)
        fbuf[(t * 16 + rb8 + r) * 64 + colc] = c[r] + bias;
    }
  }
  __syncthreads();

  // max over the 20 real edges, per column
  for (int c0 = lane; c0 < 64; c0 += 32) {
    float m = fbuf[c0];
    for (int r = 1; r < KK; ++r) m = fmaxf(m, fbuf[r * 64 + c0]);
    cm[c0] = m;
    _Float16 h = (_Float16)m;
    x1h[(size_t)p * 64 + c0]   = h;
    xcat[(size_t)p * 192 + c0] = h;
  }
  __syncthreads();
  if (lane == 0) {
    float s = 0.f;
    for (int c = 0; c < 64; ++c) {
      float v = (float)(_Float16)cm[c];   // norm of the f16-rounded features
      s += v * v;
    }
    sq1[p] = s;
  }
}

// ---------------------------------------------------------------------------
// Kernel 3: kNN in 64-D feature space — WMMA Gram tiles fused with top-20.
// one wave per (batch, 16-row tile); d = |i|^2 + |j|^2 - 2*G(i,j)
// ---------------------------------------------------------------------------
__global__ __launch_bounds__(32) void knn64_kernel(const _Float16* x1h,
                                                   const float* sq1, int* idx2) {
  __shared__ float dbuf[16 * 16];
  int wg   = blockIdx.x;               // 8*256
  int b    = wg >> 8;
  int it   = wg & 255;
  int lane = threadIdx.x;
  int rowg = (b << 12) + it * 16;      // global first row

  v16h a0 = load_a(x1h, 64, rowg, 0);
  v16h a1 = load_a(x1h, 64, rowg, 32);

  int col16 = lane & 15, rb8 = (lane >> 4) << 3;
  float sqi[8];
#pragma unroll
  for (int r = 0; r < 8; ++r) sqi[r] = sq1[rowg + rb8 + r];

  float bd[KK]; int bi[KK];
  for (int k = 0; k < KK; ++k) { bd[k] = 3.402823e38f; bi[k] = 0; }

  for (int jt = 0; jt < 256; ++jt) {
    int colg = (b << 12) + jt * 16;
    v16h q0 = load_a(x1h, 64, colg, 0);     // symmetric layout: B(k,j)=x1[j,k]
    v16h q1 = load_a(x1h, 64, colg, 32);
    v8f c = {};
    c = wmma32(a0, q0, c);
    c = wmma32(a1, q1, c);
    float sqj = sq1[colg + col16];
#pragma unroll
    for (int r = 0; r < 8; ++r)
      dbuf[(rb8 + r) * 16 + col16] = sqi[r] + sqj - 2.f * c[r];
    __syncthreads();
    if (lane < 16) {
      for (int cc = 0; cc < 16; ++cc) {
        float d = dbuf[lane * 16 + cc];
        if (d < bd[KK - 1]) {
          int p = KK - 1;
          while (p > 0 && d < bd[p - 1]) { bd[p] = bd[p - 1]; bi[p] = bi[p - 1]; --p; }
          bd[p] = d; bi[p] = jt * 16 + cc;
        }
      }
    }
    __syncthreads();
  }
  if (lane < 16) {
    size_t base = ((size_t)rowg + lane) * KK;
    for (int k = 0; k < KK; ++k) idx2[base + k] = bi[k];
  }
}

// ---------------------------------------------------------------------------
// Kernel 4: EdgeConv2 — one wave per point; linear 128->128 via WMMA,
// max over K=20, writes xcat[:,64:192] as f16.
// ---------------------------------------------------------------------------
__global__ __launch_bounds__(32) void edgeconv2_kernel(
    const _Float16* x1h, const int* idx2, const _Float16* c2w, const float* c2b,
    _Float16* xcat) {
  __shared__ __align__(32) _Float16 ebuf[32 * 128];
  __shared__ float    fbuf[32 * 128];
  int p = blockIdx.x;
  int b = p >> 12;
  int lane = threadIdx.x;

  for (int i = lane; i < 32 * 128; i += 32) ebuf[i] = (_Float16)0.f;
  __syncthreads();
  if (lane < KK) {
    int j = idx2[(size_t)p * KK + lane];
    const _Float16* xi = x1h + (size_t)p * 64;
    const _Float16* xj = x1h + (((size_t)b << 12) + j) * 64;
    _Float16* e = &ebuf[lane * 128];
    for (int c = 0; c < 64; ++c) {
      float vi = (float)xi[c];
      e[c]      = xi[c];
      e[64 + c] = (_Float16)((float)xj[c] - vi);
    }
  }
  __syncthreads();

  int col16 = lane & 15, rb8 = (lane >> 4) << 3;
  for (int t = 0; t < 2; ++t) {
    v16h a[4];
#pragma unroll
    for (int ks = 0; ks < 4; ++ks) a[ks] = load_a(ebuf, 128, t * 16, ks * 32);
    for (int nt = 0; nt < 8; ++nt) {
      v8f c = {};
#pragma unroll
      for (int ks = 0; ks < 4; ++ks)
        c = wmma32(a[ks], load_bp(c2w, nt * 4 + ks), c);
      int colc = nt * 16 + col16;
      float bias = c2b[colc];
#pragma unroll
      for (int r = 0; r < 8; ++r)
        fbuf[(t * 16 + rb8 + r) * 128 + colc] = c[r] + bias;
    }
  }
  __syncthreads();
  for (int c0 = lane; c0 < 128; c0 += 32) {
    float m = fbuf[c0];
    for (int r = 1; r < KK; ++r) m = fmaxf(m, fbuf[r * 128 + c0]);
    xcat[(size_t)p * 192 + 64 + c0] = (_Float16)m;
  }
}

// ---------------------------------------------------------------------------
// Kernel 5: lin1 (192->1024) fused with global max-pool over N.
// 128 workgroups (8 batches x 16 col-groups), 4 waves/wg, wave owns 16 cols.
// B fragments preloaded in registers; running max across all 256 row tiles.
// ---------------------------------------------------------------------------
__global__ __launch_bounds__(128) void lin1pool_kernel(
    const _Float16* xcat, const _Float16* lw, const float* lbf, float* pooled) {
  int wg   = blockIdx.x;
  int b    = wg >> 4;
  int cg   = wg & 15;
  int wave = threadIdx.x >> 5;
  int lane = threadIdx.x & 31;
  int nt   = cg * 4 + wave;            // global ntile 0..63
  int n0   = nt * 16;

  v16h bf[6];
#pragma unroll
  for (int ks = 0; ks < 6; ++ks) bf[ks] = load_bp(lw, nt * 6 + ks);

  float mx[8];
#pragma unroll
  for (int r = 0; r < 8; ++r) mx[r] = -3.402823e38f;

  for (int rt = 0; rt < 256; ++rt) {
    int row0 = (b << 12) + rt * 16;
    if (rt < 255)                       // prefetch next 16x192 f16 tile
      __builtin_prefetch(xcat + ((size_t)row0 + 16) * 192, 0, 1);
    v8f c = {};
#pragma unroll
    for (int ks = 0; ks < 6; ++ks) {
      v16h a = load_a(xcat, 192, row0, ks * 32);
      c = wmma32(a, bf[ks], c);
    }
#pragma unroll
    for (int r = 0; r < 8; ++r) mx[r] = fmaxf(mx[r], c[r]);
  }
  float m = mx[0];
#pragma unroll
  for (int r = 1; r < 8; ++r) m = fmaxf(m, mx[r]);
  m = fmaxf(m, __shfl_xor(m, 16, 32));     // combine lanes L and L+16 (same col)
  if (lane < 16) {
    int col = n0 + lane;
    pooled[b * 1024 + col] = m + lbf[col]; // bias commutes with row-max
  }
}

// ---------------------------------------------------------------------------
// Kernel 6: head MLP 1024->512->256->40 + log_softmax; one block per cloud.
// ---------------------------------------------------------------------------
__global__ __launch_bounds__(256) void head_kernel(
    const float* pooled, const float* w1, const float* b1,
    const float* w2, const float* b2, const float* w3, const float* b3,
    float* out) {
  __shared__ float in[1024];
  __shared__ float h1[512];
  __shared__ float h2[256];
  __shared__ float h3[OUTC];
  int b = blockIdx.x, t = threadIdx.x;
  for (int i = t; i < 1024; i += 256) in[i] = pooled[b * 1024 + i];
  __syncthreads();
  for (int o = t; o < 512; o += 256) {
    float s = b1[o];
    for (int k = 0; k < 1024; ++k) s += in[k] * w1[k * 512 + o];
    h1[o] = fmaxf(s, 0.f);
  }
  __syncthreads();
  if (t < 256) {
    float s = b2[t];
    for (int k = 0; k < 512; ++k) s += h1[k] * w2[k * 256 + t];
    h2[t] = fmaxf(s, 0.f);
  }
  __syncthreads();
  if (t < OUTC) {
    float s = b3[t];
    for (int k = 0; k < 256; ++k) s += h2[k] * w3[k * OUTC + t];
    h3[t] = s;
  }
  __syncthreads();
  if (t == 0) {
    float mx = h3[0];
    for (int c = 1; c < OUTC; ++c) mx = fmaxf(mx, h3[c]);
    float se = 0.f;
    for (int c = 0; c < OUTC; ++c) se += expf(h3[c] - mx);
    float l = logf(se);
    for (int c = 0; c < OUTC; ++c) out[b * OUTC + c] = h3[c] - mx - l;
  }
}

// ---------------------------------------------------------------------------
// Host-side launch
// ---------------------------------------------------------------------------
extern "C" void kernel_launch(void* const* d_in, const int* in_sizes, int n_in,
                              void* d_out, int out_size, void* d_ws, size_t ws_size,
                              hipStream_t stream) {
  const float* pos   = (const float*)d_in[0];
  // d_in[1] = batch (unused; equal-size clouds)
  const float* c1w1  = (const float*)d_in[2];
  const float* c1b1  = (const float*)d_in[3];
  const float* c1g1  = (const float*)d_in[4];
  const float* c1be1 = (const float*)d_in[5];
  const float* c1w2  = (const float*)d_in[6];
  const float* c1b2  = (const float*)d_in[7];
  const float* c1g2  = (const float*)d_in[8];
  const float* c1be2 = (const float*)d_in[9];
  const float* c1w3  = (const float*)d_in[10];
  const float* c1b3  = (const float*)d_in[11];
  const float* c2w   = (const float*)d_in[12];
  const float* c2b   = (const float*)d_in[13];
  const float* lw    = (const float*)d_in[14];
  const float* lb    = (const float*)d_in[15];
  const float* hw1   = (const float*)d_in[16];
  const float* hb1   = (const float*)d_in[17];
  const float* hw2   = (const float*)d_in[18];
  const float* hb2   = (const float*)d_in[19];
  const float* hw3   = (const float*)d_in[20];
  const float* hb3   = (const float*)d_in[21];
  float* out = (float*)d_out;

  // carve workspace (256-byte aligned slabs)
  char* base = (char*)d_ws;
  size_t off = 0;
  auto alloc = [&](size_t bytes) -> char* {
    char* p = base + off;
    off += (bytes + 255) & ~(size_t)255;
    return p;
  };
  const int P = BB * NN;                           // 32768 points
  int*      idx1   = (int*)      alloc((size_t)P * KK * 4);
  int*      idx2   = (int*)      alloc((size_t)P * KK * 4);
  _Float16* x1h    = (_Float16*) alloc((size_t)P * 64 * 2);
  float*    sq1    = (float*)    alloc((size_t)P * 4);
  _Float16* xcat   = (_Float16*) alloc((size_t)P * 192 * 2);
  float*    pooled = (float*)    alloc((size_t)BB * 1024 * 4);
  _Float16* w1p    = (_Float16*) alloc(32 * 64 * 2);
  float*    b1f    = (float*)    alloc(64 * 4);
  _Float16* w2h    = (_Float16*) alloc(64 * 64 * 2);
  float*    b2f    = (float*)    alloc(64 * 4);
  _Float16* w3h    = (_Float16*) alloc(64 * 64 * 2);
  float*    b3f    = (float*)    alloc(64 * 4);
  _Float16* c2wh   = (_Float16*) alloc(128 * 128 * 2);
  float*    c2bf   = (float*)    alloc(128 * 4);
  _Float16* lwh    = (_Float16*) alloc(192 * 1024 * 2);
  float*    lbf    = (float*)    alloc(1024 * 4);
  (void)ws_size; (void)n_in; (void)in_sizes; (void)out_size;

  // 0) fold BN + convert weights to fragment-packed f16
  {
    int total = 32 * 64 + 64 + 64 * 64 + 64 + 64 * 64 + 64 + 128 * 128 + 128 +
                192 * 1024 + 1024;
    prep_kernel<<<(total + 255) / 256, 256, 0, stream>>>(
        c1w1, c1b1, c1g1, c1be1, c1w2, c1b2, c1g2, c1be2, c1w3, c1b3,
        c2w, c2b, lw, lb,
        w1p, b1f, w2h, b2f, w3h, b3f, c2wh, c2bf, lwh, lbf);
  }
  // 1) kNN in 3D
  knn3_kernel<<<BB * (NN / 256), 256, 0, stream>>>(pos, idx1);
  // 2) EdgeConv1 (WMMA MLP + max over K)
  edgeconv1_kernel<<<P, 32, 0, stream>>>(pos, idx1, w1p, b1f, w2h, b2f, w3h, b3f,
                                         x1h, xcat, sq1);
  // 3) kNN in 64-D feature space (WMMA Gram + fused top-20)
  knn64_kernel<<<BB * (NN / 16), 32, 0, stream>>>(x1h, sq1, idx2);
  // 4) EdgeConv2 (WMMA 128x128 + max over K)
  edgeconv2_kernel<<<P, 32, 0, stream>>>(x1h, idx2, c2wh, c2bf, xcat);
  // 5) lin1 192->1024 fused with global max-pool
  lin1pool_kernel<<<BB * 16, 128, 0, stream>>>(xcat, lwh, lbf, pooled);
  // 6) head MLP + log_softmax
  head_kernel<<<BB, 256, 0, stream>>>(pooled, hw1, hb1, hw2, hb2, hw3, hb3, out);
}
